// AggrGATMean_52905407152678
// MI455X (gfx1250) — matrix operand order repaired and checked by reading, not compiled, not run on hardware
//
#include <hip/hip_runtime.h>
#include <hip/hip_bf16.h>

typedef _Float16 v16h __attribute__((ext_vector_type(16)));
typedef _Float16 v8h  __attribute__((ext_vector_type(8)));
typedef float    v8f  __attribute__((ext_vector_type(8)));
typedef float    v4f  __attribute__((ext_vector_type(4)));

#define B_   2
#define V_   50000
#define T_   3
#define N_   32
#define F_   128
#define FE_  128
#define U_   128
#define E_   300000
#define NEGF (-200.0f)
#define ALPHA_ 0.3f
#define LDA  136   // LDS row stride in halves (272B = 17*16B, staggers banks, keeps 16B align)

static __device__ __forceinline__ float redAdd16(float x) {
  x += __shfl_xor(x, 1, 32);
  x += __shfl_xor(x, 2, 32);
  x += __shfl_xor(x, 4, 32);
  x += __shfl_xor(x, 8, 32);
  return x;
}

static __device__ __forceinline__ v16h packFrag(const _Float16* p0, const _Float16* p1) {
  union { v16h v; v8h h[2]; } u;
  u.h[0] = *(const v8h*)p0;
  u.h[1] = *(const v8h*)p1;
  return u.v;
}

// ---------------------------------------------------------------------------
// 1) init: zero s2 partials, NEG_FILL atts, zero out
// ---------------------------------------------------------------------------
__global__ __launch_bounds__(256) void init_kernel(float* __restrict__ s2buf,
                                                   float* __restrict__ atts,
                                                   float* __restrict__ out) {
  size_t i = (size_t)blockIdx.x * 256 + threadIdx.x;
  const size_t TE   = (size_t)T_ * E_;
  const size_t BVTN = (size_t)B_ * V_ * T_ * N_;
  const size_t OUTN = (size_t)B_ * V_ * U_;
  if (i < TE)   s2buf[i] = 0.0f;
  if (i < BVTN) atts[i]  = NEGF;
  if (i < OUTN) out[i]   = 0.0f;
}

// ---------------------------------------------------------------------------
// 2) W_dense [T][K=FE][U] f32 -> Wt16 [T][U][K] f16 (transposed, contiguous K)
// ---------------------------------------------------------------------------
__global__ __launch_bounds__(256) void convw_kernel(const float* __restrict__ Wd,
                                                    _Float16* __restrict__ Wt16) {
  int i = blockIdx.x * 256 + threadIdx.x;
  if (i >= T_ * FE_ * U_) return;
  int t = i >> 14;        // / (128*128)
  int rem = i & 16383;
  int k = rem >> 7;
  int u = rem & 127;
  Wt16[((size_t)t * U_ + u) * FE_ + k] = (_Float16)Wd[i];
}

// ---------------------------------------------------------------------------
// Shared GEMM main loop (stage + 16 WMMAs/wave), two epilogues
// workgroup = 256 threads = 8 waves; block = 64 edges x 128 U per (t, blockIdx.x)
// wave w: m-tile = w&3 (16 edges), column half = (w>>2)*64, 4 n-tiles of 16
// W tile staged via GLOBAL_LOAD_ASYNC_TO_LDS_B128 (byte copy, ASYNCcnt);
// A tile staged through VGPRs (needs f32->f16 convert), loads marked NT so the
// 460MB edge_feats stream does not evict L2-resident features/atts/out.
// ---------------------------------------------------------------------------
#define GEMM_PROLOGUE()                                                            \
  __shared__ __align__(16) _Float16 Ash[64][LDA];                                  \
  __shared__ __align__(16) _Float16 Wsh[128][LDA];                                 \
  const int t    = blockIdx.y;                                                     \
  const int base = blockIdx.x * 64;                                                \
  const int nE   = min(64, E_ - base);                                             \
  const int tid  = threadIdx.x;                                                    \
  { /* async-stage transposed W tile: 2048 x 16B, no VGPR round-trip */            \
    unsigned ldsW = (unsigned)(uintptr_t)&Wsh[0][0];                               \
    unsigned long long gW =                                                        \
        (unsigned long long)(uintptr_t)(Wt16 + (size_t)t * U_ * FE_);              \
    _Pragma("unroll") for (int c = tid; c < 2048; c += 256) {                      \
      int u = c >> 4, kg = c & 15;                                                 \
      unsigned lds_addr = ldsW + (unsigned)((u * LDA + kg * 8) * 2);               \
      unsigned goff     = (unsigned)(c * 16);                                      \
      asm volatile("global_load_async_to_lds_b128 %0, %1, %2"                      \
                   :: "v"(lds_addr), "v"(goff), "s"(gW) : "memory");               \
    }                                                                              \
  }                                                                                \
  { /* stage A tile (f32 -> f16): 2048 x 16B, nontemporal stream */                \
    const v4f* ag = (const v4f*)(edge_feats + ((size_t)t * E_ + base) * FE_);      \
    _Pragma("unroll") for (int c = tid; c < 2048; c += 256) {                      \
      int r = c >> 5, c4 = c & 31;                                                 \
      v4f v;                                                                       \
      if (r < nE) v = __builtin_nontemporal_load(ag + (size_t)r * 32 + c4);        \
      else { v[0] = 0.f; v[1] = 0.f; v[2] = 0.f; v[3] = 0.f; }                     \
      _Float16* dst = &Ash[r][c4 * 4];                                             \
      dst[0] = (_Float16)v[0]; dst[1] = (_Float16)v[1];                            \
      dst[2] = (_Float16)v[2]; dst[3] = (_Float16)v[3];                            \
    }                                                                              \
  }                                                                                \
  asm volatile("s_wait_asynccnt 0x0" ::: "memory");                                \
  __syncthreads();                                                                 \
  const int lane    = tid & 31;                                                    \
  const int wave    = tid >> 5;                                                    \
  const int mtile   = wave & 3;                                                    \
  const int colbase = (wave >> 2) * 64;                                            \
  const int nTiles  = (nE + 15) >> 4;                                              \
  if (mtile >= nTiles) return;                                                     \
  const int m  = lane & 15;              /* A row in tile / B-C column in tile */  \
  const int ka = (lane >> 4) << 3;       /* A: +8 K for upper half-wave */         \
  const int kb = (lane >> 4) << 4;       /* unused; B loads use ka, same k order as A */        \
  v8f acc[4];                                                                      \
  _Pragma("unroll") for (int nt = 0; nt < 4; ++nt)                                 \
    _Pragma("unroll") for (int j = 0; j < 8; ++j) acc[nt][j] = 0.0f;               \
  _Pragma("unroll") for (int kk = 0; kk < FE_; kk += 32) {                         \
    v16h a = packFrag(&Ash[mtile * 16 + m][kk + ka],                               \
                      &Ash[mtile * 16 + m][kk + 16 + ka]);                         \
    _Pragma("unroll") for (int nt = 0; nt < 4; ++nt) {                             \
      int col = colbase + nt * 16 + m;                                             \
      v16h b = packFrag(&Wsh[col][kk + ka], &Wsh[col][kk + 16 + ka]);              \
      acc[nt] = __builtin_amdgcn_wmma_f32_16x16x32_f16(                            \
          false, a, false, b, (short)0, acc[nt], false, false);                    \
    }                                                                              \
  }

// 3) GEMM pass A: s2[e] += sum_col (xform[e,col]+bias[col]) * W_att[F+col]
__global__ __launch_bounds__(256) void gemm_s2_kernel(
    const float* __restrict__ edge_feats, const _Float16* __restrict__ Wt16,
    const float* __restrict__ b_dense, const float* __restrict__ W_att,
    float* __restrict__ s2buf) {
  GEMM_PROLOGUE()
  float p[8];
#pragma unroll
  for (int j = 0; j < 8; ++j) p[j] = 0.0f;
#pragma unroll
  for (int nt = 0; nt < 4; ++nt) {
    int col  = colbase + nt * 16 + m;
    float bd = b_dense[t * U_ + col];
    float w2 = W_att[F_ + col];
#pragma unroll
    for (int j = 0; j < 8; ++j) p[j] += (acc[nt][j] + bd) * w2;
  }
#pragma unroll
  for (int j = 0; j < 8; ++j) p[j] = redAdd16(p[j]);
  if (m == 0) {
    int rowbase = base + mtile * 16 + ((lane >> 4) << 3);
#pragma unroll
    for (int j = 0; j < 8; ++j)
      atomicAdd(&s2buf[(size_t)t * E_ + rowbase + j], p[j]);
  }
}

// 7) GEMM pass D: out[rowoff[e] + col] += (xform[e,col]+bias) * att[e]
__global__ __launch_bounds__(256) void gemm_scatter_kernel(
    const float* __restrict__ edge_feats, const _Float16* __restrict__ Wt16,
    const float* __restrict__ b_dense, const float* __restrict__ attg,
    const int* __restrict__ rowoff, float* __restrict__ out) {
  GEMM_PROLOGUE()
  float bd[4];
#pragma unroll
  for (int nt = 0; nt < 4; ++nt) bd[nt] = b_dense[t * U_ + colbase + nt * 16 + m];
#pragma unroll
  for (int j = 0; j < 8; ++j) {
    int rowInBlk = mtile * 16 + ((lane >> 4) << 3) + j;
    size_t ge = (size_t)t * E_ + base + rowInBlk;
    float att = attg[ge];
    int   off = rowoff[ge];
#pragma unroll
    for (int nt = 0; nt < 4; ++nt) {
      int col = colbase + nt * 16 + m;
      atomicAdd(&out[(size_t)off + col], (acc[nt][j] + bd[nt]) * att);
    }
  }
}

// ---------------------------------------------------------------------------
// 4) per-edge logit: leaky_relu(features[b,v]·w1 + s2 + b_att), scatter to atts
// ---------------------------------------------------------------------------
__global__ __launch_bounds__(256) void logit_kernel(
    const int* __restrict__ edge_idxs, const float* __restrict__ features,
    const float* __restrict__ W_att, const float* __restrict__ b_att,
    const float* __restrict__ s2buf, float* __restrict__ atts,
    int* __restrict__ rowoff) {
  int gw = blockIdx.x * 8 + (threadIdx.x >> 5);
  if (gw >= T_ * E_) return;
  int lane = threadIdx.x & 31;
  int t = gw / E_;
  const int* idx = edge_idxs + (size_t)3 * gw;
  int b = idx[0], v = idx[1], s = idx[2];
  const float* f = features + ((size_t)b * V_ + v) * F_;
  float acc = 0.0f;
#pragma unroll
  for (int i = 0; i < 4; ++i) acc += f[lane + i * 32] * W_att[lane + i * 32];
  acc += __shfl_xor(acc, 16, 32);
  acc = redAdd16(acc);
  if (lane == 0) {
    float x  = acc + s2buf[gw] + b_att[0];
    float lg = x > 0.0f ? x : ALPHA_ * x;
    atts[(((size_t)b * V_ + v) * T_ + t) * N_ + s] = lg;
    rowoff[gw] = (b * V_ + v) * U_;
  }
}

// ---------------------------------------------------------------------------
// 5) softmax over T*N = 96 per (b,v); one wave per row
// ---------------------------------------------------------------------------
__global__ __launch_bounds__(256) void softmax_kernel(float* __restrict__ atts) {
  int row = blockIdx.x * 8 + (threadIdx.x >> 5);
  if (row >= B_ * V_) return;
  int lane = threadIdx.x & 31;
  float* a = atts + (size_t)row * (T_ * N_);
  float x0 = a[lane], x1 = a[lane + 32], x2 = a[lane + 64];
  float mx = fmaxf(x0, fmaxf(x1, x2));
  mx = fmaxf(mx, __shfl_xor(mx, 16, 32));
  mx = fmaxf(mx, __shfl_xor(mx, 8, 32));
  mx = fmaxf(mx, __shfl_xor(mx, 4, 32));
  mx = fmaxf(mx, __shfl_xor(mx, 2, 32));
  mx = fmaxf(mx, __shfl_xor(mx, 1, 32));
  float e0 = __expf(x0 - mx), e1 = __expf(x1 - mx), e2 = __expf(x2 - mx);
  float s = e0 + e1 + e2;
  s += __shfl_xor(s, 16, 32);
  s = redAdd16(s);
  float inv = 1.0f / s;
  a[lane] = e0 * inv; a[lane + 32] = e1 * inv; a[lane + 64] = e2 * inv;
}

// ---------------------------------------------------------------------------
// 6) gather normalized attention back per edge (into s2buf, now free)
// ---------------------------------------------------------------------------
__global__ __launch_bounds__(256) void gather_att_kernel(
    const int* __restrict__ edge_idxs, const float* __restrict__ atts,
    float* __restrict__ attg) {
  int i = blockIdx.x * 256 + threadIdx.x;
  if (i >= T_ * E_) return;
  int t = i / E_;
  const int* idx = edge_idxs + (size_t)3 * i;
  int b = idx[0], v = idx[1], s = idx[2];
  attg[i] = atts[(((size_t)b * V_ + v) * T_ + t) * N_ + s];
}

// ---------------------------------------------------------------------------
// 8) degree = count(adjacency >= 0) over (T,N); out *= degree
//    adjacency is a 38MB single-pass stream -> nontemporal loads
// ---------------------------------------------------------------------------
__global__ __launch_bounds__(256) void degree_kernel(
    const int* __restrict__ adjacency, float* __restrict__ out) {
  int row = blockIdx.x * 8 + (threadIdx.x >> 5);
  if (row >= B_ * V_) return;
  int lane = threadIdx.x & 31;
  const int* a = adjacency + (size_t)row * (T_ * N_);
  int a0 = __builtin_nontemporal_load(a + lane);
  int a1 = __builtin_nontemporal_load(a + lane + 32);
  int a2 = __builtin_nontemporal_load(a + lane + 64);
  int c = (a0 >= 0) + (a1 >= 0) + (a2 >= 0);
  c += __shfl_xor(c, 16, 32);
  c += __shfl_xor(c, 8, 32);
  c += __shfl_xor(c, 4, 32);
  c += __shfl_xor(c, 2, 32);
  c += __shfl_xor(c, 1, 32);
  float deg = (float)c;
  float* o = out + (size_t)row * U_;
#pragma unroll
  for (int i = 0; i < 4; ++i) o[lane + i * 32] *= deg;
}

// ---------------------------------------------------------------------------
extern "C" void kernel_launch(void* const* d_in, const int* in_sizes, int n_in,
                              void* d_out, int out_size, void* d_ws, size_t ws_size,
                              hipStream_t stream) {
  const int*   adjacency  = (const int*)  d_in[0];
  const float* features   = (const float*)d_in[1];
  const int*   edge_idxs  = (const int*)  d_in[2];
  const float* edge_feats = (const float*)d_in[3];
  const float* W_dense    = (const float*)d_in[4];
  const float* b_dense    = (const float*)d_in[5];
  const float* W_att      = (const float*)d_in[6];
  const float* b_att      = (const float*)d_in[7];
  float* out = (float*)d_out;

  const size_t TE   = (size_t)T_ * E_;            // 900,000
  const size_t BVTN = (size_t)B_ * V_ * T_ * N_;  // 9,600,000

  // workspace layout (all offsets stay 16B aligned)
  float*    s2buf  = (float*)d_ws;                 // TE floats (later reused as attg)
  int*      rowoff = (int*)(s2buf + TE);           // TE ints
  float*    atts   = (float*)(rowoff + TE);        // BVTN floats
  _Float16* Wt16   = (_Float16*)(atts + BVTN);     // T*U*FE halves

  const size_t OUTN = (size_t)B_ * V_ * U_;        // 12,800,000 (largest init region)
  init_kernel<<<(unsigned)((OUTN + 255) / 256), 256, 0, stream>>>(s2buf, atts, out);

  convw_kernel<<<(T_ * FE_ * U_ + 255) / 256, 256, 0, stream>>>(W_dense, Wt16);

  dim3 ggrid((E_ + 63) / 64, T_);
  gemm_s2_kernel<<<ggrid, 256, 0, stream>>>(edge_feats, Wt16, b_dense, W_att, s2buf);

  logit_kernel<<<(unsigned)((TE + 7) / 8), 256, 0, stream>>>(
      edge_idxs, features, W_att, b_att, s2buf, atts, rowoff);

  softmax_kernel<<<(B_ * V_ + 7) / 8, 256, 0, stream>>>(atts);

  gather_att_kernel<<<(unsigned)((TE + 255) / 256), 256, 0, stream>>>(
      edge_idxs, atts, s2buf);  // s2buf now holds per-edge attention

  gemm_scatter_kernel<<<ggrid, 256, 0, stream>>>(
      edge_feats, Wt16, b_dense, s2buf, rowoff, out);

  degree_kernel<<<(B_ * V_ + 7) / 8, 256, 0, stream>>>(adjacency, out);
}
